// trajectory2seq_9783935500439
// MI455X (gfx1250) — matrix operand order repaired and checked by reading, not compiled, not run on hardware
//
#include <hip/hip_runtime.h>

// ---------------- model dims ----------------
static constexpr int HD  = 512;   // hidden
static constexpr int BB  = 128;   // batch
static constexpr int LCC = 256;   // coord length (encoder steps)
static constexpr int LTT = 64;    // txt length (decoder steps)
static constexpr int VV  = 27;    // vocab
static constexpr int H3  = 3 * HD;

// ---------------- CDNA5 WMMA types ----------------
typedef __attribute__((ext_vector_type(16))) __bf16 bf16x16;
typedef __attribute__((ext_vector_type(8)))  float  v8f;
typedef __attribute__((ext_vector_type(4)))  unsigned int u32x4;

union TileU { u32x4 q[2]; bf16x16 v; };

__device__ __forceinline__ unsigned short f2bf(float f) {
  unsigned int u = __float_as_uint(f);
  u += 0x7FFFu + ((u >> 16) & 1u);          // round-to-nearest-even
  return (unsigned short)(u >> 16);
}

// Lane loads its 16 bf16 elements as two contiguous 16B chunks (global_load_b128).
__device__ __forceinline__ bf16x16 load_tile16(const unsigned short* p0,
                                               const unsigned short* p1) {
  TileU t;
  t.q[0] = *(const u32x4*)p0;
  t.q[1] = *(const u32x4*)p1;
  return t.v;
}

__device__ __forceinline__ v8f wmma_bf16(bf16x16 a, bf16x16 b, v8f c) {
  // v_wmma_f32_16x16x32_bf16 : D = A(16x32) * B(32x16) + C(16x16 f32)
  return __builtin_amdgcn_wmma_f32_16x16x32_bf16(false, a, false, b,
                                                 (short)0, c, false, false);
}

__device__ __forceinline__ float sigm(float x)      { return 1.0f / (1.0f + __expf(-x)); }
__device__ __forceinline__ float tanh_fast(float x) { return 1.0f - 2.0f / (__expf(2.0f * x) + 1.0f); }

// ---------------- utility kernels ----------------
__global__ void f32_to_bf16_kernel(const float* __restrict__ in,
                                   unsigned short* __restrict__ out, long n) {
  long i = (long)blockIdx.x * 256 + threadIdx.x;
  if (i < n) out[i] = f2bf(in[i]);
}

__global__ void zero_u32_kernel(unsigned int* __restrict__ p, long n) {
  long i = (long)blockIdx.x * 256 + threadIdx.x;
  if (i < n) p[i] = 0u;
}

__global__ void copy_f32_kernel(float* __restrict__ dst,
                                const float* __restrict__ src, long n) {
  long i = (long)blockIdx.x * 256 + threadIdx.x;
  if (i < n) dst[i] = src[i];
}

// xs[t][b][h] = W_enc[h,0]*x[b,0,t] + W_enc[h,1]*x[b,1,t] + b_enc[h]  (bf16 out)
__global__ void encode_fc_kernel(const float* __restrict__ x,
                                 const float* __restrict__ W_enc,
                                 const float* __restrict__ b_enc,
                                 unsigned short* __restrict__ xs) {
  long i = (long)blockIdx.x * 256 + threadIdx.x;          // t*B*H + b*H + h
  long total = (long)LCC * BB * HD;
  if (i >= total) return;
  int t = (int)(i / ((long)BB * HD));
  int rem = (int)(i % ((long)BB * HD));
  int b = rem / HD;
  int h = rem % HD;
  float x0 = x[(long)b * 2 * LCC + t];
  float x1 = x[(long)b * 2 * LCC + LCC + t];
  float val = W_enc[h * 2 + 0] * x0 + W_enc[h * 2 + 1] * x1 + b_enc[h];
  xs[i] = f2bf(val);
}

// ---------------- fused GRU step (WMMA) ----------------
// One wave per 16(batch) x 16(H) tile; 6 accumulators: gi/gh for r,z,n.
// gi = x @ W1.T (+bih later), gh = h @ Whh.T (+bhh later), then gates in fp32.
__global__ __launch_bounds__(32)
void gru_step_kernel(const unsigned short* __restrict__ A1, int K1,
                     const int* __restrict__ gidx,               // optional row gather (embedding)
                     const unsigned short* __restrict__ W1,      // [3H, K1] bf16
                     const float* __restrict__ bih,              // [3H]
                     const unsigned short* __restrict__ hprev_bf,// [B, H] bf16
                     const float* __restrict__ hprev_f,          // [B, H] f32
                     const unsigned short* __restrict__ Whh,     // [3H, H] bf16
                     const float* __restrict__ bhh,              // [3H]
                     float* __restrict__ hout_f,
                     unsigned short* __restrict__ hout_bf,
                     unsigned short* __restrict__ yout, long ystride) {
  const int lane = threadIdx.x;
  const int m0 = blockIdx.x * 16;
  const int j0 = blockIdx.y * 16;
  const int lm  = lane & 15;
  const int akb = (lane >= 16) ? 16 : 0;   // A k-base: contiguous k = 16*(lane/16) + s, matching B
  const int bkb = (lane >= 16) ? 16 : 0;   // B k-base: contiguous k = 16*(lane/16) + s, matching A
  const int ncol = j0 + lm;
  const int mrow = m0 + lm;

  long arow = mrow;
  if (gidx) arow = gidx[mrow];
  const unsigned short* Ar = A1 + arow * (long)K1;

  v8f gi0 = {0,0,0,0,0,0,0,0};
  v8f gi1 = gi0, gi2 = gi0, gh0 = gi0, gh1 = gi0, gh2 = gi0;

  for (int kc = 0; kc < K1; kc += 32) {
    bf16x16 a = load_tile16(Ar + kc + akb, Ar + kc + 8 + akb);
    const unsigned short* w0 = W1 + (long)(0 * HD + ncol) * K1 + kc + bkb;
    const unsigned short* w1 = W1 + (long)(1 * HD + ncol) * K1 + kc + bkb;
    const unsigned short* w2 = W1 + (long)(2 * HD + ncol) * K1 + kc + bkb;
    gi0 = wmma_bf16(a, load_tile16(w0, w0 + 8), gi0);
    gi1 = wmma_bf16(a, load_tile16(w1, w1 + 8), gi1);
    gi2 = wmma_bf16(a, load_tile16(w2, w2 + 8), gi2);
  }
  const unsigned short* Hr = hprev_bf + (long)mrow * HD;
  for (int kc = 0; kc < HD; kc += 32) {
    bf16x16 a = load_tile16(Hr + kc + akb, Hr + kc + 8 + akb);
    const unsigned short* w0 = Whh + (long)(0 * HD + ncol) * HD + kc + bkb;
    const unsigned short* w1 = Whh + (long)(1 * HD + ncol) * HD + kc + bkb;
    const unsigned short* w2 = Whh + (long)(2 * HD + ncol) * HD + kc + bkb;
    gh0 = wmma_bf16(a, load_tile16(w0, w0 + 8), gh0);
    gh1 = wmma_bf16(a, load_tile16(w1, w1 + 8), gh1);
    gh2 = wmma_bf16(a, load_tile16(w2, w2 + 8), gh2);
  }

  const float bi0 = bih[ncol], bi1 = bih[HD + ncol], bi2 = bih[2 * HD + ncol];
  const float bh0 = bhh[ncol], bh1 = bhh[HD + ncol], bh2 = bhh[2 * HD + ncol];
  const int mbase = m0 + ((lane >= 16) ? 8 : 0);   // C/D layout: rows 0-7 / 8-15

#pragma unroll
  for (int r = 0; r < 8; ++r) {
    int m = mbase + r;
    float rr = sigm(gi0[r] + bi0 + gh0[r] + bh0);
    float zz = sigm(gi1[r] + bi1 + gh1[r] + bh1);
    float nn = tanh_fast(gi2[r] + bi2 + rr * (gh2[r] + bh2));
    long  o  = (long)m * HD + ncol;
    float hp = hprev_f[o];
    float hv = (1.0f - zz) * nn + zz * hp;
    hout_f[o] = hv;
    unsigned short hb = f2bf(hv);
    hout_bf[o] = hb;
    if (yout) yout[(long)m * ystride + ncol] = hb;
  }
}

// ---------------- generic WMMA GEMM: C = A(bf16)[M,K] @ W(bf16)[N,K].T + bias ----------------
// vmode==1: A rows are (t*B + b) time-major; store row = b*LC + t  (bi_fc -> v)
__global__ __launch_bounds__(32)
void wmma_gemm_kernel(const unsigned short* __restrict__ A,
                      const unsigned short* __restrict__ W,
                      const float* __restrict__ bias,
                      float* __restrict__ C,
                      int N, int K, int vmode) {
  const int lane = threadIdx.x;
  const int m0 = blockIdx.x * 16, n0 = blockIdx.y * 16;
  const int lm  = lane & 15;
  const int akb = (lane >= 16) ? 16 : 0;
  const int bkb = (lane >= 16) ? 16 : 0;
  const int ncol = n0 + lm;
  const unsigned short* Ar = A + (long)(m0 + lm) * K;

  v8f acc = {0,0,0,0,0,0,0,0};
  for (int kc = 0; kc < K; kc += 32) {
    bf16x16 a = load_tile16(Ar + kc + akb, Ar + kc + 8 + akb);
    const unsigned short* w = W + (long)ncol * K + kc + bkb;
    acc = wmma_bf16(a, load_tile16(w, w + 8), acc);
  }
  const float bv = bias ? bias[ncol] : 0.0f;
  const int mbase = m0 + ((lane >= 16) ? 8 : 0);
#pragma unroll
  for (int r = 0; r < 8; ++r) {
    int m = mbase + r;
    long row = vmode ? ((long)(m & (BB - 1)) * LCC + (m >> 7)) : (long)m;
    C[row * (long)N + ncol] = acc[r] + bv;
  }
}

// vn[row] = max(||v[row,:]||, eps); one wave per row
__global__ __launch_bounds__(256)
void rownorm_kernel(const float* __restrict__ v, float* __restrict__ vn, int rows) {
  int wave = threadIdx.x >> 5, lane = threadIdx.x & 31;
  int row = blockIdx.x * 8 + wave;
  if (row >= rows) return;
  const float* r = v + (long)row * HD;
  float s = 0.f;
  for (int i = lane; i < HD; i += 32) { float x = r[i]; s += x * x; }
  for (int off = 16; off; off >>= 1) s += __shfl_xor(s, off, 32);
  if (lane == 0) vn[row] = fmaxf(sqrtf(s), 1e-8f);
}

// ---------------- attention (cosine sim + softmax + weighted sum) ----------------
// one block per batch element; 256 threads (one per coord position)
__global__ __launch_bounds__(256)
void attention_kernel(const float* __restrict__ q,     // [B,H] (decoder top h)
                      const float* __restrict__ v,     // [B*LC, H]
                      const float* __restrict__ vn,    // [B*LC]
                      float* __restrict__ attn,        // [B, LC, LT]
                      int t,
                      unsigned short* __restrict__ comb) { // [B, 2H] bf16
  __shared__ float ldsq[HD];
  __shared__ float red[LCC];
  __shared__ float wsh[LCC];
  const int b = blockIdx.x, tid = threadIdx.x;

  float s = 0.f;
  for (int i = tid; i < HD; i += 256) { float x = q[(long)b * HD + i]; ldsq[i] = x; s += x * x; }
  red[tid] = s; __syncthreads();
  for (int st = 128; st; st >>= 1) { if (tid < st) red[tid] += red[tid + st]; __syncthreads(); }
  float qn = fmaxf(sqrtf(red[0]), 1e-8f);
  __syncthreads();

  const int l = tid;
  const float* vr = v + ((long)b * LCC + l) * HD;
  float dot = 0.f;
  for (int k = 0; k < HD; ++k) dot += vr[k] * ldsq[k];
  float sim = dot / (vn[(long)b * LCC + l] * qn);

  red[tid] = sim; __syncthreads();
  for (int st = 128; st; st >>= 1) { if (tid < st) red[tid] = fmaxf(red[tid], red[tid + st]); __syncthreads(); }
  float mx = red[0]; __syncthreads();
  float e = __expf(sim - mx);
  red[tid] = e; __syncthreads();
  for (int st = 128; st; st >>= 1) { if (tid < st) red[tid] += red[tid + st]; __syncthreads(); }
  float w = e / red[0];
  wsh[tid] = w;
  attn[((long)b * LCC + l) * LTT + t] = w;
  __syncthreads();

  float a0 = 0.f, a1 = 0.f;
  for (int l2 = 0; l2 < LCC; ++l2) {
    float wl = wsh[l2];
    const float* vr2 = v + ((long)b * LCC + l2) * HD;
    a0 += wl * vr2[tid];
    a1 += wl * vr2[tid + 256];
  }
  unsigned short* cb = comb + (long)b * 2 * HD;
  cb[tid]            = f2bf(ldsq[tid]);
  cb[tid + 256]      = f2bf(ldsq[tid + 256]);
  cb[HD + tid]       = f2bf(a0);
  cb[HD + tid + 256] = f2bf(a1);
}

// logits = hid @ W_out.T + b_out (V=27), write vec_out[b,t,:], argmax -> idx[b]
__global__ __launch_bounds__(32)
void logits_argmax_kernel(const float* __restrict__ hid,
                          const float* __restrict__ Wout,
                          const float* __restrict__ bout,
                          float* __restrict__ vec_out,
                          int t, int* __restrict__ idx) {
  const int b = blockIdx.x, c = threadIdx.x;
  float val = -3.4e38f;
  if (c < VV) {
    const float* hr = hid + (long)b * HD;
    const float* wr = Wout + (long)c * HD;
    float dot = 0.f;
    for (int k = 0; k < HD; ++k) dot += hr[k] * wr[k];
    dot += bout[c];
    vec_out[(long)b * LTT * VV + (long)t * VV + c] = dot;
    val = dot;
  }
  int best = c;
  for (int off = 16; off; off >>= 1) {
    float ov = __shfl_xor(val, off, 32);
    int   oi = __shfl_xor(best, off, 32);
    if (ov > val || (ov == val && oi < best)) { val = ov; best = oi; }
  }
  if (c == 0) idx[b] = best;
}

// ---------------- host orchestration ----------------
extern "C" void kernel_launch(void* const* d_in, const int* in_sizes, int n_in,
                              void* d_out, int out_size, void* d_ws, size_t ws_size,
                              hipStream_t stream) {
  (void)in_sizes; (void)n_in; (void)out_size; (void)ws_size;
  const float* x      = (const float*)d_in[0];
  const float* emb    = (const float*)d_in[2];
  const float* W_enc  = (const float*)d_in[3];
  const float* b_enc  = (const float*)d_in[4];
  const float* e0Wih  = (const float*)d_in[5];
  const float* e0Whh  = (const float*)d_in[6];
  const float* e0bih  = (const float*)d_in[7];
  const float* e0bhh  = (const float*)d_in[8];
  const float* e1Wih  = (const float*)d_in[9];
  const float* e1Whh  = (const float*)d_in[10];
  const float* e1bih  = (const float*)d_in[11];
  const float* e1bhh  = (const float*)d_in[12];
  const float* W_bi   = (const float*)d_in[13];
  const float* b_bi   = (const float*)d_in[14];
  const float* dWih   = (const float*)d_in[15];
  const float* dWhh   = (const float*)d_in[16];
  const float* dbih   = (const float*)d_in[17];
  const float* dbhh   = (const float*)d_in[18];
  const float* W_att  = (const float*)d_in[19];
  const float* b_att  = (const float*)d_in[20];
  const float* W_out  = (const float*)d_in[21];
  const float* b_out  = (const float*)d_in[22];

  float* out = (float*)d_out;
  float* vec_out_base = out;                              // [B, LT, V]
  float* hfin_base    = out + (long)BB * LTT * VV;        // [NL, B, H]
  float* attn_base    = hfin_base + 2L * BB * HD;         // [B, LC, LT]

  char* base = (char*)d_ws;
  size_t off = 0;
  auto alloc = [&](size_t bytes) -> void* {
    off = (off + 255) & ~(size_t)255;
    void* p = base + off;
    off += bytes;
    return p;
  };

  // activations
  unsigned short* xs = (unsigned short*)alloc((size_t)LCC * BB * HD * 2);      // [LC,B,H] bf16
  unsigned short* y0 = (unsigned short*)alloc((size_t)LCC * BB * 2 * HD * 2);  // [LC,B,2H] bf16
  unsigned short* y1 = (unsigned short*)alloc((size_t)LCC * BB * 2 * HD * 2);  // [LC,B,2H] bf16
  float*          vv = (float*)alloc((size_t)BB * LCC * HD * 4);               // v [B*LC, H]
  float*          vn = (float*)alloc((size_t)BB * LCC * 4);

  // bf16 weights
  unsigned short* emb_bf = (unsigned short*)alloc((size_t)VV * HD * 2);
  unsigned short* w_e0ih = (unsigned short*)alloc((size_t)2 * H3 * HD * 2);
  unsigned short* w_e0hh = (unsigned short*)alloc((size_t)2 * H3 * HD * 2);
  unsigned short* w_e1ih = (unsigned short*)alloc((size_t)2 * H3 * 2 * HD * 2);
  unsigned short* w_e1hh = (unsigned short*)alloc((size_t)2 * H3 * HD * 2);
  unsigned short* w_bi   = (unsigned short*)alloc((size_t)HD * 2 * HD * 2);
  unsigned short* w_dih  = (unsigned short*)alloc((size_t)2 * H3 * HD * 2);
  unsigned short* w_dhh  = (unsigned short*)alloc((size_t)2 * H3 * HD * 2);
  unsigned short* w_att  = (unsigned short*)alloc((size_t)HD * 2 * HD * 2);

  // hidden-state ping-pong buffers (f32 + bf16)
  float* h0f_f[2]; unsigned short* h0f_b[2];
  float* h0b_f[2]; unsigned short* h0b_b[2];
  float* h1f_f[2]; unsigned short* h1f_b[2];
  float* h1b_f[2]; unsigned short* h1b_b[2];
  float* dh0_f[2]; unsigned short* dh0_b[2];
  float* dh1_f[2]; unsigned short* dh1_b[2];
  for (int i = 0; i < 2; ++i) {
    h0f_f[i] = (float*)alloc((size_t)BB * HD * 4); h0f_b[i] = (unsigned short*)alloc((size_t)BB * HD * 2);
    h0b_f[i] = (float*)alloc((size_t)BB * HD * 4); h0b_b[i] = (unsigned short*)alloc((size_t)BB * HD * 2);
    h1f_f[i] = (float*)alloc((size_t)BB * HD * 4); h1f_b[i] = (unsigned short*)alloc((size_t)BB * HD * 2);
    h1b_f[i] = (float*)alloc((size_t)BB * HD * 4); h1b_b[i] = (unsigned short*)alloc((size_t)BB * HD * 2);
    dh0_f[i] = (float*)alloc((size_t)BB * HD * 4); dh0_b[i] = (unsigned short*)alloc((size_t)BB * HD * 2);
    dh1_f[i] = (float*)alloc((size_t)BB * HD * 4); dh1_b[i] = (unsigned short*)alloc((size_t)BB * HD * 2);
  }
  unsigned short* comb = (unsigned short*)alloc((size_t)BB * 2 * HD * 2);
  float*          hid  = (float*)alloc((size_t)BB * HD * 4);
  int*            idx  = (int*)alloc((size_t)BB * 4);

  auto conv = [&](const float* src, unsigned short* dst, long n) {
    f32_to_bf16_kernel<<<dim3((unsigned)((n + 255) / 256)), 256, 0, stream>>>(src, dst, n);
  };
  auto zero = [&](void* p, long words) {
    zero_u32_kernel<<<dim3((unsigned)((words + 255) / 256)), 256, 0, stream>>>((unsigned int*)p, words);
  };

  // ---- weight conversion (once per call; deterministic) ----
  conv(emb,   emb_bf, (long)VV * HD);
  conv(e0Wih, w_e0ih, (long)2 * H3 * HD);
  conv(e0Whh, w_e0hh, (long)2 * H3 * HD);
  conv(e1Wih, w_e1ih, (long)2 * H3 * 2 * HD);
  conv(e1Whh, w_e1hh, (long)2 * H3 * HD);
  conv(W_bi,  w_bi,   (long)HD * 2 * HD);
  conv(dWih,  w_dih,  (long)2 * H3 * HD);
  conv(dWhh,  w_dhh,  (long)2 * H3 * HD);
  conv(W_att, w_att,  (long)HD * 2 * HD);

  // ---- encode_fc ----
  {
    long n = (long)LCC * BB * HD;
    encode_fc_kernel<<<dim3((unsigned)((n + 255) / 256)), 256, 0, stream>>>(x, W_enc, b_enc, xs);
  }

  // ---- zero initial hidden states (parity 0) + decoder feedback idx ----
  zero(h0f_f[0], BB * HD); zero(h0f_b[0], BB * HD / 2);
  zero(h0b_f[0], BB * HD); zero(h0b_b[0], BB * HD / 2);
  zero(h1f_f[0], BB * HD); zero(h1f_b[0], BB * HD / 2);
  zero(h1b_f[0], BB * HD); zero(h1b_b[0], BB * HD / 2);
  zero(idx, BB);

  const dim3 gruGrid(BB / 16, HD / 16);

  // ---- encoder layer 0 (bi-GRU, fused fwd/bwd per step index) ----
  for (int s = 0; s < LCC; ++s) {
    int pin = s & 1, pout = pin ^ 1;
    int tb = LCC - 1 - s;
    gru_step_kernel<<<gruGrid, 32, 0, stream>>>(
        xs + (size_t)s * BB * HD, HD, nullptr,
        w_e0ih, e0bih, h0f_b[pin], h0f_f[pin], w_e0hh, e0bhh,
        h0f_f[pout], h0f_b[pout],
        y0 + (size_t)s * BB * 2 * HD, (long)(2 * HD));
    gru_step_kernel<<<gruGrid, 32, 0, stream>>>(
        xs + (size_t)tb * BB * HD, HD, nullptr,
        w_e0ih + (size_t)H3 * HD, e0bih + H3, h0b_b[pin], h0b_f[pin],
        w_e0hh + (size_t)H3 * HD, e0bhh + H3,
        h0b_f[pout], h0b_b[pout],
        y0 + (size_t)tb * BB * 2 * HD + HD, (long)(2 * HD));
  }
  // ---- encoder layer 1 (input = concat fwd/bwd of layer 0, K=1024) ----
  for (int s = 0; s < LCC; ++s) {
    int pin = s & 1, pout = pin ^ 1;
    int tb = LCC - 1 - s;
    gru_step_kernel<<<gruGrid, 32, 0, stream>>>(
        y0 + (size_t)s * BB * 2 * HD, 2 * HD, nullptr,
        w_e1ih, e1bih, h1f_b[pin], h1f_f[pin], w_e1hh, e1bhh,
        h1f_f[pout], h1f_b[pout],
        y1 + (size_t)s * BB * 2 * HD, (long)(2 * HD));
    gru_step_kernel<<<gruGrid, 32, 0, stream>>>(
        y0 + (size_t)tb * BB * 2 * HD, 2 * HD, nullptr,
        w_e1ih + (size_t)H3 * 2 * HD, e1bih + H3, h1b_b[pin], h1b_f[pin],
        w_e1hh + (size_t)H3 * HD, e1bhh + H3,
        h1b_f[pout], h1b_b[pout],
        y1 + (size_t)tb * BB * 2 * HD + HD, (long)(2 * HD));
  }

  // ---- bi_fc: v[b*LC+t, :] = enc_out @ W_bi.T + b_bi  (vmode row remap) ----
  wmma_gemm_kernel<<<dim3((LCC * BB) / 16, HD / 16), 32, 0, stream>>>(
      y1, w_bi, b_bi, vv, HD, 2 * HD, 1);
  rownorm_kernel<<<dim3((LCC * BB) / 8), 256, 0, stream>>>(vv, vn, LCC * BB);

  // ---- decoder: 64 greedy steps ----
  for (int t = 0; t < LTT; ++t) {
    int pin = t & 1, pout = pin ^ 1;
    const float*          h0in_f = (t == 0) ? h0f_f[0] : dh0_f[pin];  // init = enc layer0 fwd final
    const unsigned short* h0in_b = (t == 0) ? h0f_b[0] : dh0_b[pin];
    const float*          h1in_f = (t == 0) ? h0b_f[0] : dh1_f[pin];  // init = enc layer0 bwd final
    const unsigned short* h1in_b = (t == 0) ? h0b_b[0] : dh1_b[pin];

    // layer 0: input = emb[idx[b]] (gather fused into A load)
    gru_step_kernel<<<gruGrid, 32, 0, stream>>>(
        emb_bf, HD, idx,
        w_dih, dbih, h0in_b, h0in_f, w_dhh, dbhh,
        dh0_f[pout], dh0_b[pout], nullptr, 0L);
    // layer 1: input = layer-0 output
    gru_step_kernel<<<gruGrid, 32, 0, stream>>>(
        dh0_b[pout], HD, nullptr,
        w_dih + (size_t)H3 * HD, dbih + H3, h1in_b, h1in_f,
        w_dhh + (size_t)H3 * HD, dbhh + H3,
        dh1_f[pout], dh1_b[pout], nullptr, 0L);

    // attention: cosine sim, softmax, weighted sum -> comb = [q, a] (bf16)
    attention_kernel<<<dim3(BB), 256, 0, stream>>>(
        dh1_f[pout], vv, vn, attn_base, t, comb);

    // attention_fc: hid = comb @ W_att.T + b_att  (M=128, N=512, K=1024)
    wmma_gemm_kernel<<<dim3(BB / 16, HD / 16), 32, 0, stream>>>(
        comb, w_att, b_att, hid, HD, 2 * HD, 0);

    // decoder_fc + argmax feedback
    logits_argmax_kernel<<<dim3(BB), 32, 0, stream>>>(
        hid, W_out, b_out, vec_out_base, t, idx);
  }

  // ---- final decoder hidden -> h_final [NL, B, H] (final parity = 0 after 64 steps) ----
  copy_f32_kernel<<<dim3((BB * HD + 255) / 256), 256, 0, stream>>>(hfin_base, dh0_f[0], (long)BB * HD);
  copy_f32_kernel<<<dim3((BB * HD + 255) / 256), 256, 0, stream>>>(hfin_base + (long)BB * HD, dh1_f[0], (long)BB * HD);
}
